// BioSSMMixer_83906481095354
// MI455X (gfx1250) — matrix operand order repaired and checked by reading, not compiled, not run on hardware
//
#include <hip/hip_runtime.h>
#include <hip/hip_bf16.h>

// ---------------- problem constants (from setup_inputs) ----------------
#define BB   2
#define TT   1024
#define DD   1024
#define NN   16
#define MM   (BB * TT)      // 2048 rows in (b,t) order

typedef _Float16 v16h __attribute__((ext_vector_type(16)));
typedef _Float16 v8h  __attribute__((ext_vector_type(8)));
typedef float    v8f  __attribute__((ext_vector_type(8)));
typedef unsigned int u32x4 __attribute__((ext_vector_type(4)));
typedef int          i32x8 __attribute__((ext_vector_type(8)));
typedef int          i32x4 __attribute__((ext_vector_type(4)));

#if defined(__has_builtin)
#if __has_builtin(__builtin_amdgcn_tensor_load_to_lds) && __has_builtin(__builtin_amdgcn_s_wait_tensorcnt)
#define HAVE_TDM 1
#endif
#endif
#ifndef HAVE_TDM
#define HAVE_TDM 0
#endif

// ---------------- f32 -> f16 convert (activations) ----------------
__global__ void cvt_f16_kernel(const float* __restrict__ src, _Float16* __restrict__ dst, int n) {
    int i = blockIdx.x * blockDim.x + threadIdx.x;
    if (i < n) dst[i] = (_Float16)src[i];
}

// ---------------- transpose + convert weights: src[K][Nout] -> dst[Nout][K] f16 ----------------
__global__ void tcvt_kernel(const float* __restrict__ src, _Float16* __restrict__ dst,
                            int K, int Nout) {
    __shared__ float tile[32][33];
    int k0 = blockIdx.x * 32, n0 = blockIdx.y * 32;
    int tx = threadIdx.x;
    for (int ty = threadIdx.y; ty < 32; ty += 8) {
        int k = k0 + ty, n = n0 + tx;
        tile[ty][tx] = (k < K && n < Nout) ? src[k * Nout + n] : 0.0f;
    }
    __syncthreads();
    for (int ty = threadIdx.y; ty < 32; ty += 8) {
        int n = n0 + ty, k = k0 + tx;
        if (n < Nout && k < K) dst[(size_t)n * K + k] = (_Float16)tile[tx][ty];
    }
}

// ---------------- shared epilogue ----------------
template <int MODE>
__device__ __forceinline__ void gemm_epilogue(int row, int col, float v,
                                              float* __restrict__ out0,
                                              float* __restrict__ out1,
                                              _Float16* __restrict__ outh,
                                              const float* __restrict__ bias,
                                              const float* __restrict__ hsub) {
    if (MODE == 0) {
        if (col < DD) {
            out0[(size_t)row * DD + col] = v;                 // x (f32)
            outh[(size_t)row * DD + col] = (_Float16)v;       // x (f16) for dt GEMM
        } else {
            float sg = 1.0f / (1.0f + __expf(-v));
            out1[(size_t)row * DD + (col - DD)] = v * sg;     // silu(z)
        }
    } else if (MODE == 1) {
        float t = v + bias[col];
        float sp = (t > 20.0f) ? t : __logf(1.0f + __expf(t));
        out0[(size_t)row * DD + col] = sp;                    // dt = softplus(...)
    } else if (MODE == 2) {
        out0[(size_t)row * 32 + col] = v;                     // [Bm | Cm]
    } else {
        out0[(size_t)row * DD + col] = v - hsub[(size_t)row * DD + col]; // out - h
    }
}

// ---------------- TDM descriptor build + issue (2-D tile, f16 elements) ----------------
// D# per CDNA5 ISA ch.8: group0 = {flags, lds_addr, global_addr lo/hi|type},
// group1 = {mask/data_size, dims, tile dims, strides}. data_size=1 (2 bytes).
__device__ __forceinline__ void tdm_load_tile(unsigned int lds_off, const _Float16* gptr,
                                              int K, int tile_rows) {
#if HAVE_TDM
    unsigned long long ga = (unsigned long long)(uintptr_t)gptr;
    u32x4 g0;
    g0[0] = 1u;                                            // count=1, user descriptor
    g0[1] = lds_off;                                       // lds_addr (bytes)
    g0[2] = (unsigned int)ga;                              // global_addr[31:0]
    g0[3] = (unsigned int)((ga >> 32) & 0x01ffffffu) | (2u << 30); // global_addr[56:32] | type=2
    i32x8 g1;
    g1[0] = 0x10000;                                       // workgroup_mask=0, data_size=2B
    g1[1] = (int)((K & 0xffff) << 16);                     // tensor_dim0[15:0]
    g1[2] = (int)(((unsigned)K >> 16) & 0xffffu) | (int)((tile_rows & 0xffff) << 16); // dim0 hi | dim1 lo
    g1[3] = (int)(32u << 16);                              // dim1 hi=0 | tile_dim0=32
    g1[4] = tile_rows & 0xffff;                            // tile_dim1 | tile_dim2=0
    g1[5] = K;                                             // tensor_dim0_stride[31:0]
    g1[6] = 0;                                             // stride0 hi | stride1 lo
    g1[7] = 0;
    i32x4 z4 = {0, 0, 0, 0};
    i32x8 z8 = {0, 0, 0, 0, 0, 0, 0, 0};
    __builtin_amdgcn_tensor_load_to_lds(g0, g1, z4, z4, z8, 0);   // 6-arg form (clang-23)
#else
    (void)lds_off; (void)gptr; (void)K; (void)tile_rows;
#endif
}

// ---------------- LDS-staged WMMA GEMM (big GEMMs, Nout multiple of 128) ----------------
// Block = 256 threads (8 waves) computing a 64x128 tile; K staged in 32-wide slices,
// double-buffered in LDS via the Tensor Data Mover (TENSORcnt), consumed as ds_load_b128.
template <int MODE>
__global__ void gemm_wmma_lds_kernel(const _Float16* __restrict__ A,
                                     const _Float16* __restrict__ Wt,
                                     int K,
                                     float* __restrict__ out0,
                                     float* __restrict__ out1,
                                     _Float16* __restrict__ outh,
                                     const float* __restrict__ bias,
                                     const float* __restrict__ hsub) {
    __shared__ __align__(16) _Float16 sA[2][64 * 32];    //  4 KB x2
    __shared__ __align__(16) _Float16 sB[2][128 * 32];   //  8 KB x2

    const int lane = threadIdx.x & 31;
    const int wave = threadIdx.x >> 5;
    const int wm = wave >> 2;            // 0..1 -> m sub-band
    const int wn = wave & 3;             // 0..3 -> n sub-band
    const int m0 = blockIdx.x * 64;
    const int n0 = blockIdx.y * 128;
    const int lrow = lane & 15, lhi = lane >> 4;

    const _Float16* gA = A  + (size_t)m0 * K;
    const _Float16* gB = Wt + (size_t)n0 * K;

#if HAVE_TDM
    // stage slice kk into LDS buffer `buf` (issued once, wave 0)
    #define STAGE(buf, kk)                                                             \
        if (wave == 0) {                                                               \
            tdm_load_tile((unsigned int)(uintptr_t)(&sA[(buf)][0]), gA + (kk), K, 64); \
            tdm_load_tile((unsigned int)(uintptr_t)(&sB[(buf)][0]), gB + (kk), K, 128);\
        }
    #define STAGE_WAIT() if (wave == 0) { __builtin_amdgcn_s_wait_tensorcnt(0); }
#else
    // fallback: cooperative global->LDS copy by all 256 threads
    #define STAGE(buf, kk)                                                              \
        {                                                                               \
            int r = threadIdx.x >> 2, c = threadIdx.x & 3;                              \
            *(v8h*)&sA[(buf)][r * 32 + c * 8] = *(const v8h*)(gA + (size_t)r * K + (kk) + c * 8); \
            for (int it = 0; it < 2; ++it) {                                            \
                int idx = threadIdx.x + it * 256;                                       \
                int rb = idx >> 2, cb = idx & 3;                                        \
                *(v8h*)&sB[(buf)][rb * 32 + cb * 8] = *(const v8h*)(gB + (size_t)rb * K + (kk) + cb * 8); \
            }                                                                           \
        }
    #define STAGE_WAIT()
#endif

    v8f acc00 = {}, acc01 = {}, acc10 = {}, acc11 = {};

    STAGE(0, 0)
    STAGE_WAIT()
    __syncthreads();

    int buf = 0;
    for (int kk = 0; kk < K; kk += 32) {
        const int nbuf = buf ^ 1;
        if (kk + 32 < K) { STAGE(nbuf, kk + 32) }

        // fragments from LDS (layouts per ISA 7.12.2)
        const _Float16* As = &sA[buf][0];
        const _Float16* Bs = &sB[buf][0];
        const int ar = wm * 32 + lrow;
        const int br = wn * 32 + lrow;
        v16h af0, af1, bf0, bf1;
        ((v8h*)&af0)[0] = *(const v8h*)&As[ar * 32 + lhi * 8];
        ((v8h*)&af0)[1] = *(const v8h*)&As[ar * 32 + lhi * 8 + 16];
        ((v8h*)&af1)[0] = *(const v8h*)&As[(ar + 16) * 32 + lhi * 8];
        ((v8h*)&af1)[1] = *(const v8h*)&As[(ar + 16) * 32 + lhi * 8 + 16];
        ((v8h*)&bf0)[0] = *(const v8h*)&Bs[br * 32 + lhi * 8];
        ((v8h*)&bf0)[1] = *(const v8h*)&Bs[br * 32 + lhi * 8 + 16];
        ((v8h*)&bf1)[0] = *(const v8h*)&Bs[(br + 16) * 32 + lhi * 8];
        ((v8h*)&bf1)[1] = *(const v8h*)&Bs[(br + 16) * 32 + lhi * 8 + 16];

        acc00 = __builtin_amdgcn_wmma_f32_16x16x32_f16(false, af0, false, bf0, (short)0, acc00, false, false);
        acc01 = __builtin_amdgcn_wmma_f32_16x16x32_f16(false, af0, false, bf1, (short)0, acc01, false, false);
        acc10 = __builtin_amdgcn_wmma_f32_16x16x32_f16(false, af1, false, bf0, (short)0, acc10, false, false);
        acc11 = __builtin_amdgcn_wmma_f32_16x16x32_f16(false, af1, false, bf1, (short)0, acc11, false, false);

        if (kk + 32 < K) { STAGE_WAIT() }
        __syncthreads();
        buf = nbuf;
    }
    #undef STAGE
    #undef STAGE_WAIT

    #pragma unroll
    for (int half = 0; half < 4; ++half) {
        const v8f& acc = (half == 0) ? acc00 : (half == 1) ? acc01 : (half == 2) ? acc10 : acc11;
        const int mi = (half >> 1) * 16;
        const int ni = (half & 1) * 16;
        #pragma unroll
        for (int j = 0; j < 8; ++j) {
            int row = m0 + wm * 32 + mi + j + lhi * 8;
            int col = n0 + wn * 32 + ni + lrow;
            gemm_epilogue<MODE>(row, col, acc[j], out0, out1, outh, bias, hsub);
        }
    }
}

// ---------------- direct-from-global WMMA GEMM (small Nout, used for [Bm|Cm]) ----------------
template <int MODE>
__global__ void gemm_wmma_direct_kernel(const _Float16* __restrict__ A,
                                        const _Float16* __restrict__ Wt,
                                        int K, int Nout,
                                        float* __restrict__ out0,
                                        float* __restrict__ out1,
                                        _Float16* __restrict__ outh,
                                        const float* __restrict__ bias,
                                        const float* __restrict__ hsub) {
    const int lane = threadIdx.x & 31;
    const int wave = threadIdx.x >> 5;
    const int tile = blockIdx.x * (blockDim.x >> 5) + wave;
    const int ntn  = Nout >> 5;
    const int mt = tile / ntn, nt = tile % ntn;
    const int m0 = mt << 5, n0 = nt << 5;
    const int lrow = lane & 15, lhi = lane >> 4;

    v8f acc00 = {}, acc01 = {}, acc10 = {}, acc11 = {};

    const _Float16* a0 = A + (size_t)(m0 + lrow) * K + lhi * 8;
    const _Float16* a1 = A + (size_t)(m0 + 16 + lrow) * K + lhi * 8;
    const _Float16* b0 = Wt + (size_t)(n0 + lrow) * K + lhi * 8;
    const _Float16* b1 = Wt + (size_t)(n0 + 16 + lrow) * K + lhi * 8;

    for (int kk = 0; kk < K; kk += 32) {
        if (kk + 128 < K) {
            __builtin_prefetch(a0 + kk + 128, 0, 1);
            __builtin_prefetch(b0 + kk + 128, 0, 1);
        }
        v16h af0, af1, bf0, bf1;
        ((v8h*)&af0)[0] = *(const v8h*)(a0 + kk);
        ((v8h*)&af0)[1] = *(const v8h*)(a0 + kk + 16);
        ((v8h*)&af1)[0] = *(const v8h*)(a1 + kk);
        ((v8h*)&af1)[1] = *(const v8h*)(a1 + kk + 16);
        ((v8h*)&bf0)[0] = *(const v8h*)(b0 + kk);
        ((v8h*)&bf0)[1] = *(const v8h*)(b0 + kk + 16);
        ((v8h*)&bf1)[0] = *(const v8h*)(b1 + kk);
        ((v8h*)&bf1)[1] = *(const v8h*)(b1 + kk + 16);

        acc00 = __builtin_amdgcn_wmma_f32_16x16x32_f16(false, af0, false, bf0, (short)0, acc00, false, false);
        acc01 = __builtin_amdgcn_wmma_f32_16x16x32_f16(false, af0, false, bf1, (short)0, acc01, false, false);
        acc10 = __builtin_amdgcn_wmma_f32_16x16x32_f16(false, af1, false, bf0, (short)0, acc10, false, false);
        acc11 = __builtin_amdgcn_wmma_f32_16x16x32_f16(false, af1, false, bf1, (short)0, acc11, false, false);
    }

    #pragma unroll
    for (int half = 0; half < 4; ++half) {
        const v8f& acc = (half == 0) ? acc00 : (half == 1) ? acc01 : (half == 2) ? acc10 : acc11;
        const int mi = (half >> 1) * 16;
        const int ni = (half & 1) * 16;
        #pragma unroll
        for (int j = 0; j < 8; ++j) {
            int row = m0 + mi + j + lhi * 8;
            int col = n0 + ni + lrow;
            gemm_epilogue<MODE>(row, col, acc[j], out0, out1, outh, bias, hsub);
        }
    }
}

// ---------------- sequential SSM scan + spike gate ----------------
__global__ void scan_kernel(const float* __restrict__ dt,
                            const float* __restrict__ xf,
                            const float* __restrict__ zs,
                            const float* __restrict__ BC,
                            const float* __restrict__ A_log,
                            const float* __restrict__ D_skip,
                            const float* __restrict__ v_th,
                            _Float16* __restrict__ uh) {
    const int d = blockIdx.x * blockDim.x + threadIdx.x;   // [0, DD)
    const int b = blockIdx.y;                              // [0, BB)

    float a[NN], s[NN];
    #pragma unroll
    for (int n = 0; n < NN; ++n) {
        a[n] = -__expf(A_log[d * NN + n]);
        s[n] = 0.0f;
    }
    const float vth = fmaxf(v_th[d], 0.1f);
    const float dsk = D_skip[d];

    __shared__ float sB[NN];
    __shared__ float sC[NN];

    for (int t = 0; t < TT; ++t) {
        const size_t row = (size_t)b * TT + t;
        __syncthreads();
        if (threadIdx.x < 32) {
            float v = BC[row * 32 + threadIdx.x];
            if (threadIdx.x < NN) sB[threadIdx.x] = v;
            else                  sC[threadIdx.x - NN] = v;
        }
        __syncthreads();

        const float dtv = dt[row * DD + d];
        const float xv  = xf[row * DD + d];
        const float dtx = dtv * xv;
        float y = 0.0f;
        #pragma unroll
        for (int n = 0; n < NN; ++n) {
            float dec = __expf(dtv * a[n]);
            s[n] = dec * s[n] + dtx * sB[n];
            y += s[n] * sC[n];
        }
        y += dsk * xv;
        float sp = 1.0f / (1.0f + __expf(-10.0f * (y - vth)));   // surrogate spike gate
        float u = y * sp * zs[row * DD + d];                      // * silu(z)
        uh[row * DD + d] = (_Float16)u;
    }
}

// ---------------- launcher ----------------
extern "C" void kernel_launch(void* const* d_in, const int* in_sizes, int n_in,
                              void* d_out, int out_size, void* d_ws, size_t ws_size,
                              hipStream_t stream) {
    const float* h      = (const float*)d_in[0];   // (B,T,D)
    const float* W_xz   = (const float*)d_in[1];   // (D,2D)
    const float* W_dt   = (const float*)d_in[2];   // (D,D)
    const float* b_dt   = (const float*)d_in[3];   // (D)
    const float* A_log  = (const float*)d_in[4];   // (D,N)
    const float* W_B    = (const float*)d_in[5];   // (D,N)
    const float* W_C    = (const float*)d_in[6];   // (D,N)
    const float* D_skip = (const float*)d_in[7];   // (D)
    const float* W_out  = (const float*)d_in[8];   // (D,D)
    const float* v_th   = (const float*)d_in[9];   // (D)
    float* out = (float*)d_out;                    // (B,T,D)

    char* ws = (char*)d_ws;
    const size_t MiB = 1024 * 1024;
    _Float16* hh    = (_Float16*)(ws + 0 * MiB);   // M*D f16       (4 MiB)
    _Float16* WxzT  = (_Float16*)(ws + 4 * MiB);   // 2D x D f16    (4 MiB)
    _Float16* WdtT  = (_Float16*)(ws + 8 * MiB);   // D x D f16     (2 MiB)
    _Float16* WoutT = (_Float16*)(ws + 10 * MiB);  // D x D f16     (2 MiB)
    _Float16* WbcT  = (_Float16*)(ws + 12 * MiB);  // 32 x D f16    (64 KiB)
    float*    xf    = (float*)   (ws + 13 * MiB);  // M*D f32       (8 MiB)
    _Float16* xh    = (_Float16*)(ws + 21 * MiB);  // M*D f16       (4 MiB)
    float*    zsil  = (float*)   (ws + 25 * MiB);  // M*D f32       (8 MiB)
    float*    dtb   = (float*)   (ws + 33 * MiB);  // M*D f32       (8 MiB)
    float*    BC    = (float*)   (ws + 41 * MiB);  // M*32 f32      (256 KiB)
    _Float16* uhb   = (_Float16*)(ws + 42 * MiB);  // M*D f16       (4 MiB)

    // 1) activations -> f16
    cvt_f16_kernel<<<(MM * DD) / 256, 256, 0, stream>>>(h, hh, MM * DD);

    // 2) weights: transpose + f16 convert
    dim3 tb(32, 8);
    tcvt_kernel<<<dim3(DD / 32, (2 * DD) / 32), tb, 0, stream>>>(W_xz, WxzT, DD, 2 * DD);
    tcvt_kernel<<<dim3(DD / 32, DD / 32), tb, 0, stream>>>(W_dt, WdtT, DD, DD);
    tcvt_kernel<<<dim3(DD / 32, DD / 32), tb, 0, stream>>>(W_out, WoutT, DD, DD);
    tcvt_kernel<<<dim3(DD / 32, 1), tb, 0, stream>>>(W_B, WbcT, DD, NN);                    // rows 0..15
    tcvt_kernel<<<dim3(DD / 32, 1), tb, 0, stream>>>(W_C, WbcT + (size_t)NN * DD, DD, NN);  // rows 16..31

    // 3) GEMM1: xz = h @ W_xz -> x (f32+f16), silu(z)   [TDM/LDS staged]
    gemm_wmma_lds_kernel<0><<<dim3(MM / 64, (2 * DD) / 128), 256, 0, stream>>>(
        hh, WxzT, DD, xf, zsil, xh, nullptr, nullptr);
    // 4) GEMM2: dt = softplus(x @ W_dt + b_dt)          [TDM/LDS staged]
    gemm_wmma_lds_kernel<1><<<dim3(MM / 64, DD / 128), 256, 0, stream>>>(
        xh, WdtT, DD, dtb, nullptr, nullptr, b_dt, nullptr);
    // 5) GEMM3: [Bm|Cm] = h @ [W_B|W_C]                 [direct, Nout=32]
    gemm_wmma_direct_kernel<2><<<(MM / 32) / 8, 256, 0, stream>>>(
        hh, WbcT, DD, 32, BC, nullptr, nullptr, nullptr, nullptr);
    // 6) sequential scan + spike/silu gating -> u (f16)
    scan_kernel<<<dim3(DD / 256, BB), 256, 0, stream>>>(
        dtb, xf, zsil, BC, A_log, D_skip, v_th, uhb);
    // 7) GEMM4: out = u @ W_out - h                     [TDM/LDS staged]
    gemm_wmma_lds_kernel<3><<<dim3(MM / 64, DD / 128), 256, 0, stream>>>(
        uhb, WoutT, DD, out, nullptr, nullptr, nullptr, h);
}
